// ContextModel_38560216384104
// MI455X (gfx1250) — compile-verified, matrix operand order checked by reading, not run on hardware
//
#include <hip/hip_runtime.h>
#include <hip/hip_bf16.h>

// ---------------------------------------------------------------------------
// Types for CDNA5 WMMA
// ---------------------------------------------------------------------------
typedef __attribute__((ext_vector_type(16))) __bf16         v16bf;
typedef __attribute__((ext_vector_type(8)))  float          v8f;
typedef __attribute__((ext_vector_type(8)))  unsigned short v8us;
typedef __attribute__((ext_vector_type(16))) unsigned short v16us;

#define RREL   6
#define HEADS  4
#define CDIM   64
#define HC     256   // HEADS*CDIM
#define DIN    768
#define ODIM   64
#define XLXR   512   // fused [xl | xr] row width

// ---------------------------------------------------------------------------
// helpers
// ---------------------------------------------------------------------------
__device__ __forceinline__ unsigned short f2bf(float f) {
  unsigned int u = __float_as_uint(f);
  u += 0x7FFFu + ((u >> 16) & 1u);     // round-to-nearest-even
  return (unsigned short)(u >> 16);
}

__device__ __forceinline__ void atomicMaxF(float* addr, float val) {
  if (val >= 0.0f) atomicMax((int*)addr, __float_as_int(val));
  else             atomicMin((unsigned int*)addr, __float_as_uint(val));
}

// LDS byte offset of a shared-memory pointer: generic shared addresses on
// AMDGPU are {aperture_hi, lds_offset}, so the low 32 bits are the offset.
__device__ __forceinline__ unsigned lds_off(const void* p) {
  return (unsigned)(uintptr_t)p;
}

// gfx1250 async global->LDS 16-byte copy (tracked by ASYNCcnt)
__device__ __forceinline__ void async_b128(unsigned ldsoff, const void* gptr) {
  asm volatile("global_load_async_to_lds_b128 %0, %1, off"
               :: "v"(ldsoff), "v"(gptr) : "memory");
}
__device__ __forceinline__ void wait_asynccnt0() {
  asm volatile("s_wait_asynccnt 0" ::: "memory");
}

__device__ __forceinline__ v16bf frag16(const unsigned short* base, int khalf) {
  v8us lo = *(const v8us*)(base + khalf * 8);
  v8us hi = *(const v8us*)(base + 16 + khalf * 8);
  return __builtin_bit_cast(v16bf, __builtin_shufflevector(
             lo, hi, 0,1,2,3,4,5,6,7,8,9,10,11,12,13,14,15));
}

// ---------------------------------------------------------------------------
// Big projection GEMM (bf16 inputs, double-buffered async-LDS staging):
//   C[M x 512] = Abf (M x K bf16, row-major) * Btbf^T (Btbf is [512][K] bf16)
// Block: 256 thr = 8 waves; block tile = 256 rows x 64 cols;
// wave tile = 32 rows x 64 cols -> 8 x v_wmma_f32_16x16x32_bf16 per K-step.
// Tile k+1 is issued with global_load_async_to_lds_b128 into the alternate
// LDS buffer before computing tile k, overlapping copy with WMMA.
// ---------------------------------------------------------------------------
__global__ __launch_bounds__(256) void gemm_proj_async_wmma(
    const unsigned short* __restrict__ Abf,
    const unsigned short* __restrict__ Btbf,
    float* __restrict__ C, int ldc, int M, int K)
{
  __shared__ __align__(16) unsigned short As[2][256][40];  // 2 x (256 rows x 32 k)
  __shared__ __align__(16) unsigned short Bt[2][64][40];   // 2 x (64 cols x 32 k)

  const int tid  = threadIdx.x;
  const int r0   = blockIdx.x * 256;
  const int cb   = blockIdx.y * 64;
  const int wave = tid >> 5;
  const int lane = tid & 31;
  const int l16  = lane & 15;
  const int khalf = lane >> 4;

  // staging assignments (constant per thread)
  const int sArow = tid >> 2;            // A: 4 chunks/thread, rows tid>>2 + 64*i
  const int sAch  = tid & 3;
  const int sBcol = tid >> 2;            // B: 1 chunk/thread
  const int sBch  = tid & 3;

  v8f acc[2][4];
#pragma unroll
  for (int m = 0; m < 2; ++m)
#pragma unroll
    for (int j = 0; j < 4; ++j) acc[m][j] = (v8f){};

  // issue tile 0 into buffer 0
  {
#pragma unroll
    for (int i = 0; i < 4; ++i) {
      int row = sArow + 64 * i;
      int grow = r0 + row; if (grow >= M) grow = M - 1;
      async_b128(lds_off(&As[0][row][sAch * 8]),
                 Abf + (long)grow * K + sAch * 8);
    }
    async_b128(lds_off(&Bt[0][sBcol][sBch * 8]),
               Btbf + (long)(cb + sBcol) * K + sBch * 8);
  }

  int p = 0;
  for (int k0 = 0; k0 < K; k0 += 32, p ^= 1) {
    wait_asynccnt0();
    __syncthreads();            // buffer p ready; buffer 1-p free for reuse

    const int k1 = k0 + 32;
    if (k1 < K) {               // prefetch next tile into buffer 1-p
#pragma unroll
      for (int i = 0; i < 4; ++i) {
        int row = sArow + 64 * i;
        int grow = r0 + row; if (grow >= M) grow = M - 1;
        async_b128(lds_off(&As[1 - p][row][sAch * 8]),
                   Abf + (long)grow * K + k1 + sAch * 8);
      }
      async_b128(lds_off(&Bt[1 - p][sBcol][sBch * 8]),
                 Btbf + (long)(cb + sBcol) * K + k1 + sBch * 8);
    }

#pragma unroll
    for (int m = 0; m < 2; ++m) {
      const int arow = wave * 32 + m * 16 + l16;
      v16bf a = frag16(&As[p][arow][0], khalf);
#pragma unroll
      for (int j = 0; j < 4; ++j) {
        v16bf b = frag16(&Bt[p][j * 16 + l16][0], khalf);
        acc[m][j] = __builtin_amdgcn_wmma_f32_16x16x32_bf16(
            false, a, false, b, (short)0, acc[m][j], false, false);
      }
    }
  }

  // epilogue: one base pointer per (m,j) tile, rows rbase..rbase+7 stride ldc
  const bool full = (r0 + 256 <= M);
#pragma unroll
  for (int m = 0; m < 2; ++m) {
    const int rbase = r0 + wave * 32 + m * 16 + ((lane < 16) ? 0 : 8);
#pragma unroll
    for (int j = 0; j < 4; ++j) {
      float* cp = C + (long)rbase * ldc + (cb + j * 16 + l16);
#pragma unroll
      for (int i = 0; i < 8; ++i) {
        if (full || rbase + i < M) cp[(long)i * ldc] = acc[m][j][i];
      }
    }
  }
}

// ---------------------------------------------------------------------------
// Small fused GEMM (f32 inputs converted to bf16 while staging into LDS):
//   C = [A1 | A2 split at kSplit] * [B1 ; B2 split at kSplit] (+bias, relu)
// Block tile 128 rows x 64 cols; wave tile 16 x 64 (4 WMMAs / K-step).
// ---------------------------------------------------------------------------
__global__ __launch_bounds__(256) void gemm_bf16_wmma(
    const float* __restrict__ A1, const float* __restrict__ A2, int kSplit,
    int lda1, int lda2,
    const float* __restrict__ B1, const float* __restrict__ B2,
    int ldb1, int ldb2,
    float* __restrict__ C, int ldc,
    const float* __restrict__ bias, int relu,
    int M, int K)
{
  __shared__ __align__(16) unsigned short As[128][40];
  __shared__ __align__(16) unsigned short Bt[64][40];

  const int tid  = threadIdx.x;
  const int r0   = blockIdx.x * 128;
  const int cb   = blockIdx.y * 64;
  const int wave = tid >> 5;
  const int lane = tid & 31;
  const int l16  = lane & 15;
  const int khalf = lane >> 4;

  v8f acc0 = {}, acc1 = {}, acc2 = {}, acc3 = {};

  for (int k0 = 0; k0 < K; k0 += 32) {
    const float* Ab; int lda; int ka;
    const float* Bb; int ldb; int kb;
    if (k0 < kSplit) { Ab = A1; lda = lda1; ka = k0;          Bb = B1; ldb = ldb1; kb = k0; }
    else             { Ab = A2; lda = lda2; ka = k0 - kSplit; Bb = B2; ldb = ldb2; kb = k0 - kSplit; }

    __syncthreads();
    {
      int row  = tid >> 1;
      int kk   = (tid & 1) * 16;
      int grow = r0 + row; if (grow >= M) grow = M - 1;
      const float* src = Ab + (long)grow * lda + ka + kk;
#pragma unroll
      for (int i = 0; i < 16; ++i) As[row][kk + i] = f2bf(src[i]);
    }
    {
      int kk = tid >> 3;
      int cg = (tid & 7) * 8;
      const float* src = Bb + (long)(kb + kk) * ldb + cb + cg;
#pragma unroll
      for (int i = 0; i < 8; ++i) Bt[cg + i][kk] = f2bf(src[i]);
    }
    __syncthreads();

    const int arow = wave * 16 + l16;
    v16bf a = frag16(&As[arow][0], khalf);
#pragma unroll
    for (int j = 0; j < 4; ++j) {
      v16bf b = frag16(&Bt[j * 16 + l16][0], khalf);
      if      (j == 0) acc0 = __builtin_amdgcn_wmma_f32_16x16x32_bf16(false, a, false, b, (short)0, acc0, false, false);
      else if (j == 1) acc1 = __builtin_amdgcn_wmma_f32_16x16x32_bf16(false, a, false, b, (short)0, acc1, false, false);
      else if (j == 2) acc2 = __builtin_amdgcn_wmma_f32_16x16x32_bf16(false, a, false, b, (short)0, acc2, false, false);
      else             acc3 = __builtin_amdgcn_wmma_f32_16x16x32_bf16(false, a, false, b, (short)0, acc3, false, false);
    }
  }

  const bool full = (r0 + 128 <= M);
  const int rbase = r0 + wave * 16 + ((lane < 16) ? 0 : 8);
#pragma unroll
  for (int j = 0; j < 4; ++j) {
    const int col = cb + j * 16 + l16;
    const float bv = bias ? bias[col] : 0.0f;
    float* cp = C + (long)rbase * ldc + col;
    v8f acc = (j == 0) ? acc0 : (j == 1) ? acc1 : (j == 2) ? acc2 : acc3;
#pragma unroll
    for (int i = 0; i < 8; ++i) {
      if (full || rbase + i < M) {
        float v = acc[i] + bv;
        if (relu) v = fmaxf(v, 0.0f);
        cp[(long)i * ldc] = v;
      }
    }
  }
}

// ---------------------------------------------------------------------------
// conversion kernels
// ---------------------------------------------------------------------------
__global__ void convert_x_kernel(const float* __restrict__ x,
                                 unsigned short* __restrict__ xb, long n) {
  long i = (long)blockIdx.x * blockDim.x + threadIdx.x;
  if (i < n) xb[i] = f2bf(x[i]);
}

// WT[c][k] = bf16( c<256 ? Wl[k][c] : Wr[k][c-256] ), WT is [512][768]
__global__ void convert_wt_kernel(const float* __restrict__ Wl,
                                  const float* __restrict__ Wr,
                                  unsigned short* __restrict__ WT) {
  int idx = blockIdx.x * blockDim.x + threadIdx.x;
  if (idx >= XLXR * DIN) return;
  int c = idx & (XLXR - 1);
  int k = idx >> 9;
  float v = (c < HC) ? Wl[(long)k * HC + c] : Wr[(long)k * HC + (c - HC)];
  WT[(long)c * DIN + k] = f2bf(v);
}

__global__ void fill_kernel(float* __restrict__ p, long n, float v) {
  long i = (long)blockIdx.x * blockDim.x + threadIdx.x;
  if (i < n) p[i] = v;
}

// ---------------------------------------------------------------------------
// GATv2 edge logits (warp per edge; E real edges + N self-loops)
// xlxr rows are [xl(256) | xr(256)]
// ---------------------------------------------------------------------------
__global__ void edge_logits_kernel(
    const float* __restrict__ xlxr,
    const int* __restrict__ src, const int* __restrict__ tgt,
    int E, int nE, const float* __restrict__ att,
    float* __restrict__ logits, float* __restrict__ m)
{
  int e    = (int)(((long)blockIdx.x * blockDim.x + threadIdx.x) >> 5);
  int lane = threadIdx.x & 31;
  if (e >= nE) return;
  int s, t;
  if (e < E) { s = src[e]; t = tgt[e]; } else { s = e - E; t = s; }
  const float* pl = xlxr + (long)s * XLXR;
  const float* pr = xlxr + (long)t * XLXR + HC;
  float part[4] = {0.f, 0.f, 0.f, 0.f};
#pragma unroll
  for (int i = 0; i < 8; ++i) {
    int c   = lane + i * 32;
    float v = pl[c] + pr[c];
    v = (v > 0.f) ? v : 0.2f * v;
    part[i >> 1] += v * att[c];
  }
#pragma unroll
  for (int off = 16; off > 0; off >>= 1) {
    part[0] += __shfl_xor(part[0], off, 32);
    part[1] += __shfl_xor(part[1], off, 32);
    part[2] += __shfl_xor(part[2], off, 32);
    part[3] += __shfl_xor(part[3], off, 32);
  }
  if (lane == 0) {
#pragma unroll
    for (int h = 0; h < HEADS; ++h) {
      logits[(long)e * HEADS + h] = part[h];
      atomicMaxF(&m[(long)t * HEADS + h], part[h]);
    }
  }
}

__global__ void edge_exp_kernel(
    float* __restrict__ logits, const float* __restrict__ m,
    const int* __restrict__ tgt, int E, int nE,
    float* __restrict__ denom)
{
  long idx = (long)blockIdx.x * blockDim.x + threadIdx.x;
  if (idx >= (long)nE * HEADS) return;
  int e = (int)(idx >> 2), h = (int)(idx & 3);
  int t = (e < E) ? tgt[e] : (e - E);
  float a = __expf(logits[idx] - m[(long)t * HEADS + h]);
  logits[idx] = a;
  atomicAdd(&denom[(long)t * HEADS + h], a);
}

// gat[t][c] += xl[s][c] * alpha[e][c>>6]   (warp per edge)
__global__ void edge_msg_kernel(
    const float* __restrict__ xlxr, const float* __restrict__ a,
    const float* __restrict__ denom,
    const int* __restrict__ src, const int* __restrict__ tgt,
    int E, int nE, float* __restrict__ gat)
{
  int e    = (int)(((long)blockIdx.x * blockDim.x + threadIdx.x) >> 5);
  int lane = threadIdx.x & 31;
  if (e >= nE) return;
  int s, t;
  if (e < E) { s = src[e]; t = tgt[e]; } else { s = e - E; t = s; }
  float al[4];
#pragma unroll
  for (int h = 0; h < HEADS; ++h)
    al[h] = a[(long)e * HEADS + h] / (denom[(long)t * HEADS + h] + 1e-16f);
  const float* pl = xlxr + (long)s * XLXR;
  float* pg = gat + (long)t * HC;
  for (int c = lane; c < HC; c += 32)
    atomicAdd(&pg[c], pl[c] * al[c >> 6]);
}

__global__ void bias_relu_kernel(float* __restrict__ g, const float* __restrict__ b, long n) {
  long i = (long)blockIdx.x * blockDim.x + threadIdx.x;
  if (i < n) g[i] = fmaxf(g[i] + b[i & (HC - 1)], 0.0f);
}

// agg[t][*] += h[s][*]   (warp per real edge)
__global__ void edge_agg_kernel(
    const float* __restrict__ h, const int* __restrict__ src,
    const int* __restrict__ tgt, int E, int width, float* __restrict__ agg)
{
  int e    = (int)(((long)blockIdx.x * blockDim.x + threadIdx.x) >> 5);
  int lane = threadIdx.x & 31;
  if (e >= E) return;
  int s = src[e], t = tgt[e];
  const float* ph = h + (long)s * width;
  float* pa = agg + (long)t * width;
  for (int c = lane; c < width; c += 32) atomicAdd(&pa[c], ph[c]);
}

// ---------------------------------------------------------------------------
// host-side launcher
// ---------------------------------------------------------------------------
extern "C" void kernel_launch(void* const* d_in, const int* in_sizes, int n_in,
                              void* d_out, int out_size, void* d_ws, size_t ws_size,
                              hipStream_t stream) {
  const float* x       = (const float*)d_in[0];
  const int*   ei      = (const int*)  d_in[1];
  const float* Wl      = (const float*)d_in[2];
  const float* Wr      = (const float*)d_in[3];
  const float* att     = (const float*)d_in[4];
  const float* b_gat   = (const float*)d_in[5];
  const float* W_rel1  = (const float*)d_in[6];
  const float* W_root1 = (const float*)d_in[7];
  const float* b1      = (const float*)d_in[8];
  const float* W_rel2  = (const float*)d_in[9];
  const float* W_root2 = (const float*)d_in[10];
  const float* b2      = (const float*)d_in[11];
  float* out = (float*)d_out;

  const int N  = in_sizes[0] / DIN;
  const int E  = in_sizes[1] / (RREL * 2);
  const int nE = E + N;

  // workspace layout
  unsigned short* xb = (unsigned short*)d_ws;               // [N][768] bf16
  unsigned short* WT = xb + (size_t)N * DIN;                // [512][768] bf16
  float* xlxr  = (float*)(WT + (size_t)XLXR * DIN);         // [N][512]; later overlaid
  float* gat   = xlxr + (size_t)N * XLXR;                   // [N][256] gat -> h
  float* av    = gat + (size_t)N * HC;                      // [nE][4] logits -> a
  float* m     = av + (size_t)nE * HEADS;                   // [N][4]
  float* denom = m + (size_t)N * HEADS;                     // [N][4]
  // overlays (xlxr is dead after edge_msg):
  float* agg   = xlxr;                                      // [N][256]
  float* h1    = xlxr + (size_t)N * HC;                     // [N][64]
  float* agg2  = xlxr + (size_t)N * (HC + ODIM);            // [N][64]

  const dim3 blk(256);
  const int gRows128 = (N + 127) / 128;
  const int gRows256 = (N + 255) / 256;
  const int gWarp  = (nE + 7) / 8;
  const int gWarpE = (E + 7) / 8;

  // x -> bf16 once
  long nx = (long)N * DIN;
  hipLaunchKernelGGL(convert_x_kernel, dim3((nx + 255) / 256), blk, 0, stream, x, xb, nx);

  for (int r = 0; r < RREL; ++r) {
    const int* src = ei + (size_t)r * 2 * E;
    const int* tgt = src + E;

    // weights -> bf16, transposed [512][768]
    hipLaunchKernelGGL(convert_wt_kernel, dim3((XLXR * DIN + 255) / 256), blk, 0, stream,
        Wl + (size_t)r * DIN * HC, Wr + (size_t)r * DIN * HC, WT);

    // [xl | xr] = x @ [Wl | Wr]  (double-buffered async-LDS bf16 WMMA)
    hipLaunchKernelGGL(gemm_proj_async_wmma, dim3(gRows256, XLXR / 64), blk, 0, stream,
        xb, WT, xlxr, XLXR, N, DIN);

    // softmax prep
    long nm = (long)N * HEADS;
    hipLaunchKernelGGL(fill_kernel, dim3((nm + 255) / 256), blk, 0, stream, m, nm, -3.0e38f);
    hipLaunchKernelGGL(fill_kernel, dim3((nm + 255) / 256), blk, 0, stream, denom, nm, 0.0f);

    hipLaunchKernelGGL(edge_logits_kernel, dim3(gWarp), blk, 0, stream,
        xlxr, src, tgt, E, nE, att + (size_t)r * HC, av, m);
    long na = (long)nE * HEADS;
    hipLaunchKernelGGL(edge_exp_kernel, dim3((na + 255) / 256), blk, 0, stream,
        av, m, tgt, E, nE, denom);

    // gat accumulation
    long ng = (long)N * HC;
    hipLaunchKernelGGL(fill_kernel, dim3((ng + 255) / 256), blk, 0, stream, gat, ng, 0.0f);
    hipLaunchKernelGGL(edge_msg_kernel, dim3(gWarp), blk, 0, stream,
        xlxr, av, denom, src, tgt, E, nE, gat);
    hipLaunchKernelGGL(bias_relu_kernel, dim3((ng + 255) / 256), blk, 0, stream,
        gat, b_gat + (size_t)r * HC, ng);

    // GraphConv1: agg (overlays xlxr) + fused GEMM [agg|h] @ [W_rel1;W_root1] + b1, relu
    hipLaunchKernelGGL(fill_kernel, dim3((ng + 255) / 256), blk, 0, stream, agg, ng, 0.0f);
    hipLaunchKernelGGL(edge_agg_kernel, dim3(gWarpE), blk, 0, stream, gat, src, tgt, E, HC, agg);
    hipLaunchKernelGGL(gemm_bf16_wmma, dim3(gRows128, 1), blk, 0, stream,
        agg, gat, HC, HC, HC,
        W_rel1 + (size_t)r * HC * CDIM, W_root1 + (size_t)r * HC * CDIM, CDIM, CDIM,
        h1, CDIM, b1 + (size_t)r * CDIM, 1, N, 2 * HC);

    // GraphConv2: agg2 + fused GEMM [agg2|h1] @ [W_rel2;W_root2] + b2 -> out[:, r, :]
    long n2 = (long)N * ODIM;
    hipLaunchKernelGGL(fill_kernel, dim3((n2 + 255) / 256), blk, 0, stream, agg2, n2, 0.0f);
    hipLaunchKernelGGL(edge_agg_kernel, dim3(gWarpE), blk, 0, stream, h1, src, tgt, E, ODIM, agg2);
    hipLaunchKernelGGL(gemm_bf16_wmma, dim3(gRows128, 1), blk, 0, stream,
        agg2, h1, CDIM, CDIM, CDIM,
        W_rel2 + (size_t)r * CDIM * ODIM, W_root2 + (size_t)r * CDIM * ODIM, ODIM, ODIM,
        out + (size_t)r * ODIM, RREL * ODIM, b2 + (size_t)r * ODIM, 0, N, 2 * CDIM);
  }
  (void)n_in; (void)out_size; (void)ws_size;
}
